// BAAFNet_60610578481726
// MI455X (gfx1250) — matrix operand order repaired and checked by reading, not compiled, not run on hardware
//
#include <hip/hip_runtime.h>
#include <hip/hip_bf16.h>

typedef __attribute__((ext_vector_type(16))) _Float16 v16h;
typedef __attribute__((ext_vector_type(8)))  _Float16 v8h;
typedef __attribute__((ext_vector_type(8)))  float    v8f;

#define BB   2
#define NN   4096
#define KK   16
#define BN   (BB * NN)        // 8192 points
#define BNK  (BN * KK)        // 131072 neighbor rows

struct MlpArgs {
    const float* W;    // (O, I) row-major, f32 (as given)
    const float* b;    // (O)
    const float* g;    // BN gamma or nullptr
    const float* beta;
    const float* m;
    const float* v;
};

// ---------------------------------------------------------------------------
// KNN: per-thread scan of all 4096 candidates (staged through LDS),
// keep top-16 LARGEST d2 (reference uses jax.lax.top_k on d2).
// ---------------------------------------------------------------------------
__global__ __launch_bounds__(256) void knn_kernel(const float* __restrict__ p,
                                                  int* __restrict__ idx) {
    __shared__ float sp[256 * 3];
    const int t   = threadIdx.x;
    const int pid = blockIdx.x * 256 + t;           // block shares batch b
    const int b   = pid / NN;
    const float px = p[(size_t)pid * 3 + 0];
    const float py = p[(size_t)pid * 3 + 1];
    const float pz = p[(size_t)pid * 3 + 2];

    float vals[KK];
    int   ids[KK];
#pragma unroll
    for (int i = 0; i < KK; ++i) { vals[i] = -1e30f; ids[i] = 0; }

    for (int base = 0; base < NN; base += 256) {
        __syncthreads();
        const size_t src = ((size_t)b * NN + base + t) * 3;
        sp[t * 3 + 0] = p[src + 0];
        sp[t * 3 + 1] = p[src + 1];
        sp[t * 3 + 2] = p[src + 2];
        __syncthreads();
        for (int j = 0; j < 256; ++j) {
            const float dx = sp[j * 3 + 0] - px;
            const float dy = sp[j * 3 + 1] - py;
            const float dz = sp[j * 3 + 2] - pz;
            const float d2 = dx * dx + dy * dy + dz * dz;
            if (d2 > vals[KK - 1]) {
                int pos = KK - 1;
                while (pos > 0 && d2 > vals[pos - 1]) {
                    vals[pos] = vals[pos - 1]; ids[pos] = ids[pos - 1]; --pos;
                }
                vals[pos] = d2; ids[pos] = base + j;
            }
        }
    }
#pragma unroll
    for (int i = 0; i < KK; ++i) idx[(size_t)pid * KK + i] = ids[i];
}

// ---------------------------------------------------------------------------
// f32 -> f16 conversion (weights, once per launch)
// ---------------------------------------------------------------------------
__global__ __launch_bounds__(256) void cvt_f16_kernel(const float* __restrict__ s,
                                                      _Float16* __restrict__ d, int n) {
    const int i = blockIdx.x * 256 + threadIdx.x;
    if (i < n) d[i] = (_Float16)s[i];
}

// ---------------------------------------------------------------------------
// Fused MLP via WMMA f32_16x16x32_f16, f16 operands in memory.
//   Y[row, colOff+col] = act(BN(concat(X0,X1)[row,:] . W[col,:] + b[col]))
// Block = 128 threads = 4 waves = 4 M-tiles; each wave computes NT N-tiles,
// reusing its A fragment NT times. All operand loads are 16B (v8h) vectors.
// K-loop unrolled x2 with explicit ping-pong buffers so next-chunk loads
// overlap current-chunk WMMAs without register rotation copies.
// I % 32 == 0, O % (16*NT) == 0, M % 64 == 0, split % 8 == 0.
// ACT: 0=none, 1=relu, 2=lrelu(0.2)
// ---------------------------------------------------------------------------
__device__ __forceinline__ v8h loadA8(const _Float16* __restrict__ X0,
                                      const _Float16* __restrict__ X1,
                                      int split, int I, size_t row, int k8) {
    if (k8 < split) return *(const v8h*)(X0 + row * (size_t)split + k8);
    return *(const v8h*)(X1 + row * (size_t)(I - split) + (k8 - split));
}

template <int ACT, int NT, bool OUT_HALF>
__global__ __launch_bounds__(128) void mlp_wmma_kernel(
    const _Float16* __restrict__ X0, const _Float16* __restrict__ X1, int split,
    int I, MlpArgs P, const _Float16* __restrict__ Wh,
    float* __restrict__ Yf, _Float16* __restrict__ Yh, int ldY, int colOff) {
    const int lane   = threadIdx.x & 31;
    const int wid    = threadIdx.x >> 5;
    const int mtile  = blockIdx.x * 4 + wid;
    const int ntbase = blockIdx.y * NT;
    const int lane15 = lane & 15;
    const int hiA    = (lane >= 16) ? 8 : 0;   // A: K offset for high half-wave
    const int hiB    = (lane >= 16) ? 8 : 0;   // B: K offset for high half-wave
    const size_t mrow = (size_t)mtile * 16 + lane15;

    int col[NT];
#pragma unroll
    for (int t = 0; t < NT; ++t) col[t] = (ntbase + t) * 16 + lane15;

    // hoist epilogue coefficients so their loads overlap the GEMM loop
    const bool bn = (P.g != nullptr);
    float bias[NT], scale[NT], shift[NT];
#pragma unroll
    for (int t = 0; t < NT; ++t) {
        bias[t] = P.b[col[t]];
        scale[t] = 1.f; shift[t] = 0.f;
        if (bn) {
            const float inv = rsqrtf(P.v[col[t]] + 1e-5f);
            scale[t] = P.g[col[t]] * inv;
            shift[t] = P.beta[col[t]] - P.m[col[t]] * scale[t];
        }
    }

    v8f acc[NT];
#pragma unroll
    for (int t = 0; t < NT; ++t) acc[t] = v8f{};

    // ping-pong fragment buffers (no rotation: each phase uses its own regs)
    v8h a0A, a1A, w0A[NT], w1A[NT];
    v8h a0B, a1B, w0B[NT], w1B[NT];

    auto loadChunk = [&](v8h& a0, v8h& a1, v8h (&w0)[NT], v8h (&w1)[NT], int kbase) {
        a0 = loadA8(X0, X1, split, I, mrow, kbase + hiA);
        a1 = loadA8(X0, X1, split, I, mrow, kbase + hiA + 16);
#pragma unroll
        for (int t = 0; t < NT; ++t) {
            const _Float16* wp = Wh + (size_t)col[t] * I + kbase + hiB;
            w0[t] = *(const v8h*)wp;
            w1[t] = *(const v8h*)(wp + 16);
        }
    };
    auto doMma = [&](const v8h& a0, const v8h& a1,
                     const v8h (&w0)[NT], const v8h (&w1)[NT]) {
        const v16h a = __builtin_shufflevector(a0, a1, 0, 1, 2, 3, 4, 5, 6, 7,
                                               8, 9, 10, 11, 12, 13, 14, 15);
#pragma unroll
        for (int t = 0; t < NT; ++t) {
            const v16h w = __builtin_shufflevector(w0[t], w1[t],
                                                   0, 1, 2, 3, 4, 5, 6, 7,
                                                   8, 9, 10, 11, 12, 13, 14, 15);
            acc[t] = __builtin_amdgcn_wmma_f32_16x16x32_f16(
                false, a, false, w, (short)0, acc[t], false, false);
        }
    };

    loadChunk(a0A, a1A, w0A, w1A, 0);
    int kb = 0;
    while (true) {
        int kn = kb + 32;
        if (kn < I) loadChunk(a0B, a1B, w0B, w1B, kn);   // overlap phase-A WMMAs
        doMma(a0A, a1A, w0A, w1A);
        kb = kn;
        if (kb >= I) break;

        kn = kb + 32;
        if (kn < I) loadChunk(a0A, a1A, w0A, w1A, kn);   // overlap phase-B WMMAs
        doMma(a0B, a1B, w0B, w1B);
        kb = kn;
        if (kb >= I) break;
    }

    // epilogue: bias + folded BN + activation
    const int r0 = (lane >= 16) ? 8 : 0;
#pragma unroll
    for (int t = 0; t < NT; ++t) {
#pragma unroll
        for (int r = 0; r < 8; ++r) {
            const size_t row = (size_t)mtile * 16 + r0 + r;
            float y = acc[t][r] + bias[t];
            if (bn) y = y * scale[t] + shift[t];
            if (ACT == 1) y = fmaxf(y, 0.f);
            else if (ACT == 2) y = (y > 0.f) ? y : 0.2f * y;
            const size_t o = row * (size_t)ldY + colOff + col[t];
            if (OUT_HALF) Yh[o] = (_Float16)y;
            else          Yf[o] = y;
        }
    }
}

// ---------------------------------------------------------------------------
// mlp0: 6 -> 32, BN + lrelu, scalar (I=6 too skinny for WMMA); f16 out
// ---------------------------------------------------------------------------
__global__ __launch_bounds__(256) void mlp0_kernel(const float* __restrict__ X,
                                                   MlpArgs P,
                                                   _Float16* __restrict__ Y) {
    const int t = blockIdx.x * blockDim.x + threadIdx.x;
    if (t >= BN) return;
    float x[6];
#pragma unroll
    for (int i = 0; i < 6; ++i) x[i] = X[(size_t)t * 6 + i];
#pragma unroll
    for (int o = 0; o < 32; ++o) {
        float acc = P.b[o];
#pragma unroll
        for (int i = 0; i < 6; ++i) acc = fmaf(x[i], P.W[o * 6 + i], acc);
        const float inv = rsqrtf(P.v[o] + 1e-5f);
        acc = (acc - P.m[o]) * P.g[o] * inv + P.beta[o];
        Y[(size_t)t * 32 + o] = (_Float16)((acc > 0.f) ? acc : 0.2f * acc);
    }
}

// ---------------------------------------------------------------------------
// gather + embed: lgc (6, f32) and lsc (2*dh, f16) per (b,n,k)
// ---------------------------------------------------------------------------
__global__ __launch_bounds__(256) void gather_embed_kernel(
    const float* __restrict__ p, const float* __restrict__ f0,
    const int* __restrict__ idx, float* __restrict__ lgc,
    _Float16* __restrict__ lsc, int dh) {
    const size_t t = (size_t)blockIdx.x * blockDim.x + threadIdx.x;
    if (t >= (size_t)BNK) return;
    const size_t bn = t / KK;
    const int b = (int)(bn / NN), n = (int)(bn % NN);
    const int j = idx[t];
    const size_t pe = ((size_t)b * NN + n) * 3, pk = ((size_t)b * NN + j) * 3;
    float* lg = lgc + t * 6;
#pragma unroll
    for (int c = 0; c < 3; ++c) {
        const float a = p[pe + c];
        lg[c] = a; lg[3 + c] = p[pk + c] - a;
    }
    const float* fe = f0 + ((size_t)b * NN + n) * dh;
    const float* fk = f0 + ((size_t)b * NN + j) * dh;
    _Float16* ls = lsc + t * (size_t)(2 * dh);
    for (int c = 0; c < dh; ++c) {
        const float a = fe[c];
        ls[c]      = (_Float16)a;
        ls[dh + c] = (_Float16)(fk[c] - a);
    }
}

// ---------------------------------------------------------------------------
// ba1 (d->3, BN+relu) + p_knn residual; write lgc_aug = [lgc(6), p_tilde(3)]
// ---------------------------------------------------------------------------
__global__ __launch_bounds__(256) void ba1_kernel(
    const _Float16* __restrict__ lsc, const float* __restrict__ p,
    const int* __restrict__ idx, const float* __restrict__ lgc,
    MlpArgs P, float* __restrict__ lgc_aug, int d) {
    const size_t t = (size_t)blockIdx.x * blockDim.x + threadIdx.x;
    if (t >= (size_t)BNK) return;
    const size_t bn = t / KK;
    const int b = (int)(bn / NN);
    const int j = idx[t];
    const _Float16* row = lsc + t * (size_t)d;
    float* la = lgc_aug + t * 9;
#pragma unroll
    for (int c = 0; c < 6; ++c) la[c] = lgc[t * 6 + c];
#pragma unroll
    for (int o = 0; o < 3; ++o) {
        float acc = P.b[o];
        for (int i = 0; i < d; ++i) acc = fmaf((float)row[i], P.W[o * d + i], acc);
        const float inv = rsqrtf(P.v[o] + 1e-5f);
        acc = (acc - P.m[o]) * P.g[o] * inv + P.beta[o];
        acc = fmaxf(acc, 0.f);
        la[6 + o] = acc + p[((size_t)b * NN + j) * 3 + o];
    }
}

// ---------------------------------------------------------------------------
// ba2 (9->dh, +f_knn residual, f16 out) and ba3 (9->dh -> alc[:,0:dh], f16)
// ---------------------------------------------------------------------------
__global__ __launch_bounds__(256) void ba23_kernel(
    const float* __restrict__ lgc_aug, const float* __restrict__ f0,
    const int* __restrict__ idx, MlpArgs P2, MlpArgs P3,
    _Float16* __restrict__ f_tilde, _Float16* __restrict__ alc, int dh) {
    const size_t t = (size_t)blockIdx.x * blockDim.x + threadIdx.x;
    if (t >= (size_t)BNK) return;
    const size_t bn = t / KK;
    const int b = (int)(bn / NN);
    const int j = idx[t];
    float x[9];
#pragma unroll
    for (int i = 0; i < 9; ++i) x[i] = lgc_aug[t * 9 + i];
    const float* fk = f0 + ((size_t)b * NN + j) * dh;
    const int d = 2 * dh;
    for (int c = 0; c < dh; ++c) {
        float a2 = P2.b[c], a3 = P3.b[c];
#pragma unroll
        for (int i = 0; i < 9; ++i) {
            a2 = fmaf(x[i], P2.W[c * 9 + i], a2);
            a3 = fmaf(x[i], P3.W[c * 9 + i], a3);
        }
        a2 = (a2 - P2.m[c]) * P2.g[c] * rsqrtf(P2.v[c] + 1e-5f) + P2.beta[c];
        a3 = (a3 - P3.m[c]) * P3.g[c] * rsqrtf(P3.v[c] + 1e-5f) + P3.beta[c];
        f_tilde[t * (size_t)dh + c] = (_Float16)(fmaxf(a2, 0.f) + fk[c]);
        alc[t * (size_t)d + c]      = (_Float16)fmaxf(a3, 0.f);
    }
}

// ---------------------------------------------------------------------------
// softmax over K (per channel) -> weighted sum; max over K; concat (f16 out)
// ---------------------------------------------------------------------------
__global__ __launch_bounds__(256) void pool_kernel(
    const _Float16* __restrict__ alc, const float* __restrict__ logits,
    _Float16* __restrict__ mlain, int d) {
    const size_t t = (size_t)blockIdx.x * blockDim.x + threadIdx.x;
    if (t >= (size_t)BN * d) return;
    const int c = (int)(t % d);
    const size_t bn = t / d;
    const _Float16* al = alc + bn * (size_t)KK * d + c;
    const float*    lg = logits + bn * (size_t)KK * d + c;
    float mxl = -1e30f;
#pragma unroll
    for (int k = 0; k < KK; ++k) mxl = fmaxf(mxl, lg[(size_t)k * d]);
    float se = 0.f, ws = 0.f, mxa = -1e30f;
#pragma unroll
    for (int k = 0; k < KK; ++k) {
        const float e = __expf(lg[(size_t)k * d] - mxl);
        se += e;
        const float a = (float)al[(size_t)k * d];
        ws = fmaf(a, e, ws);
        mxa = fmaxf(mxa, a);
    }
    mlain[bn * (size_t)(2 * d) + c]     = (_Float16)(ws / se);
    mlain[bn * (size_t)(2 * d) + d + c] = (_Float16)mxa;
}

// ---------------------------------------------------------------------------
// Host-side orchestration
// ---------------------------------------------------------------------------
static MlpArgs take6(void* const* d, int& i) {
    // jax pytree sorted-key order: W, b, beta, g, m, v
    MlpArgs a;
    a.W    = (const float*)d[i++];
    a.b    = (const float*)d[i++];
    a.beta = (const float*)d[i++];
    a.g    = (const float*)d[i++];
    a.m    = (const float*)d[i++];
    a.v    = (const float*)d[i++];
    return a;
}
static MlpArgs take2(void* const* d, int& i) {
    MlpArgs a = {};
    a.W = (const float*)d[i++];
    a.b = (const float*)d[i++];
    return a;
}

template <int ACT, bool OUT_HALF>
static void launch_mlp(int M, int I, int split, int O,
                       const _Float16* X0, const _Float16* X1,
                       MlpArgs P, const _Float16* Wh,
                       float* Yf, _Float16* Yh, int ldY, int colOff,
                       hipStream_t s) {
    if (O % 64 == 0)
        mlp_wmma_kernel<ACT, 4, OUT_HALF><<<dim3(M / 64, O / 64), 128, 0, s>>>(
            X0, X1, split, I, P, Wh, Yf, Yh, ldY, colOff);
    else
        mlp_wmma_kernel<ACT, 2, OUT_HALF><<<dim3(M / 64, O / 32), 128, 0, s>>>(
            X0, X1, split, I, P, Wh, Yf, Yh, ldY, colOff);
}

extern "C" void kernel_launch(void* const* d_in, const int* in_sizes, int n_in,
                              void* d_out, int out_size, void* d_ws, size_t ws_size,
                              hipStream_t stream) {
    const float* p = (const float*)d_in[0];
    const float* f = (const float*)d_in[1];

    int ai = 2; // pytree order: 'layers' (list) before 'mlp0'
    MlpArgs L[3][8]; // ba0..ba4, mla0, mla1, mla2
    for (int l = 0; l < 3; ++l) {
        for (int j = 0; j < 5; ++j) L[l][j] = take6(d_in, ai); // ba0..ba4
        L[l][5] = take2(d_in, ai);                             // mla0 (bn=False)
        L[l][6] = take6(d_in, ai);                             // mla1
        L[l][7] = take6(d_in, ai);                             // mla2
    }
    MlpArgs mlp0 = take6(d_in, ai);

    // workspace carve (max-layer sizes, 256B aligned)
    char* w = (char*)d_ws;
    auto carve = [&](size_t bytes) {
        char* r = w;
        w += (bytes + 255) & ~(size_t)255;
        return r;
    };
    int*      idx     = (int*)     carve((size_t)BNK * 4);
    _Float16* fA      = (_Float16*)carve((size_t)BN * 512 * 2);
    _Float16* fB      = (_Float16*)carve((size_t)BN * 512 * 2);
    float*    f0      = (float*)   carve((size_t)BN * 128 * 4);
    _Float16* lsc     = (_Float16*)carve((size_t)BNK * 256 * 2);
    float*    lgc     = (float*)   carve((size_t)BNK * 6 * 4);
    float*    lgc_aug = (float*)   carve((size_t)BNK * 9 * 4);
    _Float16* f_tilde = (_Float16*)carve((size_t)BNK * 128 * 2);
    _Float16* alc     = (_Float16*)carve((size_t)BNK * 256 * 2);
    float*    logits  = (float*)   carve((size_t)BNK * 256 * 4);
    _Float16* mlain   = (_Float16*)carve((size_t)BN * 512 * 2);
    _Float16* t1      = (_Float16*)carve((size_t)BN * 256 * 2);
    // f16 weight staging (reused per layer)
    _Float16* Wba0  = (_Float16*)carve((size_t)128 * 256 * 2);
    _Float16* Wba4  = (_Float16*)carve((size_t)128 * 384 * 2);
    _Float16* Wmla0 = (_Float16*)carve((size_t)256 * 256 * 2);
    _Float16* Wmla1 = (_Float16*)carve((size_t)256 * 512 * 2);
    _Float16* Wmla2 = (_Float16*)carve((size_t)512 * 256 * 2);

    const int ew = 256;
    auto eg = [&](size_t total) { return (unsigned)((total + ew - 1) / ew); };
    auto cvt = [&](const float* s, _Float16* d, int n) {
        cvt_f16_kernel<<<eg(n), ew, 0, stream>>>(s, d, n);
    };

    // KNN once (p never changes across layers)
    knn_kernel<<<BN / 256, 256, 0, stream>>>(p, idx);

    // mlp0: 6 -> 32, BN + lrelu -> f16
    mlp0_kernel<<<eg(BN), ew, 0, stream>>>(f, mlp0, fA);

    const int d_in_arr[3] = {32, 128, 256};
    const int d_arr[3]    = {64, 128, 256};
    _Float16* fcur = fA;
    _Float16* fnxt = fB;

    for (int l = 0; l < 3; ++l) {
        const int dI = d_in_arr[l];
        const int d  = d_arr[l];
        const int dh = d / 2;

        // stage this layer's GEMM weights as f16
        cvt(L[l][0].W, Wba0,  dh * dI);
        cvt(L[l][4].W, Wba4,  dh * (d + dh));
        cvt(L[l][5].W, Wmla0, d * d);
        cvt(L[l][6].W, Wmla1, d * 2 * d);
        cvt(L[l][7].W, Wmla2, 2 * d * d);

        // ba0: (BN, dI) -> (BN, dh) f32, BN+relu — WMMA
        launch_mlp<1, false>(BN, dI, dI, dh, fcur, nullptr, L[l][0], Wba0,
                             f0, nullptr, dh, 0, stream);

        // gather + embed -> lgc (f32), lsc (f16)
        gather_embed_kernel<<<eg(BNK), ew, 0, stream>>>(p, f0, idx, lgc, lsc, dh);

        // ba1 + p_knn residual -> lgc_aug (9, f32)
        ba1_kernel<<<eg(BNK), ew, 0, stream>>>(lsc, p, idx, lgc, L[l][1], lgc_aug, d);

        // ba2 (+f_knn -> f_tilde f16) and ba3 (-> alc[:,0:dh] f16)
        ba23_kernel<<<eg(BNK), ew, 0, stream>>>(lgc_aug, f0, idx, L[l][2], L[l][3],
                                                f_tilde, alc, dh);

        // ba4: concat(lsc[d], f_tilde[dh]) -> alc[:, dh:d] f16, BN+relu — WMMA
        launch_mlp<1, true>(BNK, d + dh, d, dh, lsc, f_tilde, L[l][4], Wba4,
                            nullptr, alc, d, dh, stream);

        // mla0: (BNK, d) -> (BNK, d) f32 logits, bias only — WMMA
        launch_mlp<0, false>(BNK, d, d, d, alc, nullptr, L[l][5], Wmla0,
                             logits, nullptr, d, 0, stream);

        // softmax-weighted sum + max pool over K -> (BN, 2d) f16
        pool_kernel<<<eg((size_t)BN * d), ew, 0, stream>>>(alc, logits, mlain, d);

        // mla1: (BN, 2d) -> (BN, d) f16, BN+relu — WMMA
        launch_mlp<1, true>(BN, 2 * d, 2 * d, d, mlain, nullptr, L[l][6], Wmla1,
                            nullptr, t1, d, 0, stream);

        // mla2: (BN, d) -> (BN, 2d), BN+lrelu — WMMA
        if (l == 2) {
            // final layer: f32 straight to d_out
            launch_mlp<2, false>(BN, d, d, 2 * d, t1, nullptr, L[l][7], Wmla2,
                                 (float*)d_out, nullptr, 2 * d, 0, stream);
        } else {
            launch_mlp<2, true>(BN, d, d, 2 * d, t1, nullptr, L[l][7], Wmla2,
                                nullptr, fnxt, 2 * d, 0, stream);
            _Float16* tmp = fcur; fcur = fnxt; fnxt = tmp;
        }
    }
}
